// SpatialScaledDotProductAttention_386547057192
// MI455X (gfx1250) — matrix operand order repaired and checked by reading, not compiled, not run on hardware
//
#include <hip/hip_runtime.h>

// Shapes from the reference
#define B_  16
#define H_  8
#define T_  12
#define N_  325
#define DK_ 32

constexpr int ROW_TILES = 21;   // ceil(325/16)
constexpr int COL_TILES = 21;   // ceil(325/16) -> 336 padded score cols
constexpr int PV_CHUNKS = 11;   // ceil(325/32) -> 352 padded prob cols
constexpr int PCOLS     = 352;
constexpr int PVEC      = PCOLS / 4;   // 88 float4 per row
constexpr int SC_STRIDE = 356;  // floats; rows 16B-aligned -> ds b128 ops
constexpr float NEG_BIG = -1e30f;

typedef _Float16 v16h __attribute__((ext_vector_type(16)));
typedef float    v8f  __attribute__((ext_vector_type(8)));
typedef float    v4f  __attribute__((ext_vector_type(4)));

static __device__ __forceinline__ int imin(int a, int b) { return a < b ? a : b; }

__global__ __launch_bounds__(32)
void attn_wmma_kernel(const float* __restrict__ Q,
                      const float* __restrict__ K,
                      const float* __restrict__ V,
                      const float* __restrict__ bias,
                      float* __restrict__ out)
{
    __shared__ float sc[16 * SC_STRIDE];   // scores -> exp(S - m) (in place)
    __shared__ float sinv[16];             // per-row 1/sum

    const int wg   = blockIdx.x;
    const int rt   = wg % ROW_TILES;       // 16-row tile within problem
    const int bht  = wg / ROW_TILES;       // flattened (b,h,t)
    const int bh   = bht / T_;             // b*H + h (bias has no T dim)
    const int row0 = rt * 16;

    const int lane = threadIdx.x;          // 0..31 (wave32)
    const int g    = lane >> 4;            // 16-lane group
    const int ln   = lane & 15;

    const float scale = 0.17677669529663687f;  // 1/sqrt(32), folded into Q

    const float* Qp = Q    + (size_t)bht * N_ * DK_;
    const float* Kp = K    + (size_t)bht * N_ * DK_;
    const float* Vp = V    + (size_t)bht * N_ * DK_;
    const float* Bp = bias + (size_t)bh  * N_ * N_;
    float*       Op = out  + (size_t)bht * N_ * DK_;

    // ---- pad score cols 336..351 with -1e30 (cols 325..335 handled in pass 1)
    {
        const int prow = lane >> 1;                 // 2 lanes per row
        const int pcol = 336 + (lane & 1) * 8;      // 8 floats each
        v4f negv = { NEG_BIG, NEG_BIG, NEG_BIG, NEG_BIG };
        *(v4f*)(sc + prow * SC_STRIDE + pcol)     = negv;
        *(v4f*)(sc + prow * SC_STRIDE + pcol + 4) = negv;
    }

    // ---- Q tile as A-fragment (16x32 f16), scale folded in --------------
    // A k order used here (contiguous, not the ISA table order): group g, elem h -> K = 16g+h
    // Rows >= N clamped to row N-1 (their outputs are never stored).
    v16h aq;
    {
        const int   qrow = imin(row0 + ln, N_ - 1);
        const v4f*  q4   = (const v4f*)(Qp + qrow * DK_);
        v4f q0 = q4[4 * g];         // K = 16g .. 16g+3
        v4f q1 = q4[4 * g + 1];     // K = 16g+4 .. 16g+7
        v4f q2 = q4[4 * g + 2];     // K = 16g+8 .. 16g+11
        v4f q3 = q4[4 * g + 3];     // K = 16g+12 .. 16g+15
        #pragma unroll
        for (int i = 0; i < 4; ++i) {
            aq[i]      = (_Float16)(q0[i] * scale);
            aq[4 + i]  = (_Float16)(q1[i] * scale);
            aq[8 + i]  = (_Float16)(q2[i] * scale);
            aq[12 + i] = (_Float16)(q3[i] * scale);
        }
    }

    // ---- pass 1: scores = (Q*scale) K^T + bias -> LDS -------------------
    for (int j = 0; j < COL_TILES; ++j) {
        const int  col   = j * 16 + ln;        // score column = K row
        const int  ck    = imin(col, N_ - 1);  // clamp for safe load
        const bool valid = col < N_;

        // B (32x16 f16), same k order as A: lane%16 = n, elem h -> K-dim = 16g + h
        const v4f* k4 = (const v4f*)(Kp + ck * DK_);
        v4f k0 = k4[4 * g];
        v4f k1 = k4[4 * g + 1];
        v4f k2 = k4[4 * g + 2];
        v4f k3 = k4[4 * g + 3];
        v16h bk;
        #pragma unroll
        for (int i = 0; i < 4; ++i) {
            bk[i]      = (_Float16)k0[i];
            bk[4 + i]  = (_Float16)k1[i];
            bk[8 + i]  = (_Float16)k2[i];
            bk[12 + i] = (_Float16)k3[i];
        }

        // Unconditional bias loads from clamped addresses (overlap with WMMA)
        float bvals[8];
        #pragma unroll
        for (int r = 0; r < 8; ++r) {
            const int cr = imin(row0 + r + 8 * g, N_ - 1);
            bvals[r] = Bp[(size_t)cr * N_ + ck];
        }

        v8f c = {};
        c = __builtin_amdgcn_wmma_f32_16x16x32_f16(false, aq, false, bk,
                                                   (short)0, c, false, false);

        // f32 C/D layout: VGPR r, lane -> row = r + 8g, col = lane%16.
        // Invalid cols get -1e30 so the vectorized softmax needs no tail logic.
        #pragma unroll
        for (int r = 0; r < 8; ++r) {
            const float s = valid ? (c[r] + bvals[r]) : NEG_BIG;
            sc[(r + 8 * g) * SC_STRIDE + j * 16 + ln] = s;
        }
    }
    __syncthreads();

    // ---- pass 2: row max + exp + sum (2 lanes per row, b128 vectorized) -
    // Padding cols hold -1e30 -> exp() = 0. Probs stay UNNORMALIZED in LDS;
    // 1/sum goes to sinv[] and is applied to the final accumulators instead
    // (normalization commutes through the PV matmul).
    {
        const int row  = lane >> 1;
        const int part = lane & 1;
        v4f* srow = (v4f*)(sc + row * SC_STRIDE);

        v4f mv = { NEG_BIG, NEG_BIG, NEG_BIG, NEG_BIG };
        #pragma unroll 4
        for (int i = part; i < PVEC; i += 2) {
            const v4f x = srow[i];
            #pragma unroll
            for (int k = 0; k < 4; ++k) mv[k] = fmaxf(mv[k], x[k]);
        }
        float m = fmaxf(fmaxf(mv[0], mv[1]), fmaxf(mv[2], mv[3]));
        m = fmaxf(m, __shfl_xor(m, 1, 32));

        float s = 0.0f;
        #pragma unroll 4
        for (int i = part; i < PVEC; i += 2) {
            v4f x = srow[i];
            v4f e;
            #pragma unroll
            for (int k = 0; k < 4; ++k) e[k] = __expf(x[k] - m);
            s += (e[0] + e[1]) + (e[2] + e[3]);
            srow[i] = e;                       // store unnormalized exp
        }
        s += __shfl_xor(s, 1, 32);

        if (part == 0) sinv[row] = 1.0f / s;   // exec restored after the if
    }
    __syncthreads();

    // ---- pass 3: context = exp(S-m) @ V (K=32 chunks, two DK-halves) ----
    v8f acc0 = {};
    v8f acc1 = {};
    for (int ch = 0; ch < PV_CHUNKS; ++ch) {
        const int m0 = ch * 32;

        // A = P chunk 16x32: row = lane%16, K = prob column (b128 LDS loads)
        const v4f* p4 = (const v4f*)(sc + ln * SC_STRIDE + m0);
        v4f p0 = p4[4 * g];
        v4f p1 = p4[4 * g + 1];
        v4f p2 = p4[4 * g + 2];
        v4f p3 = p4[4 * g + 3];
        v16h ap;
        #pragma unroll
        for (int i = 0; i < 4; ++i) {
            ap[i]      = (_Float16)p0[i];
            ap[4 + i]  = (_Float16)p1[i];
            ap[8 + i]  = (_Float16)p2[i];
            ap[12 + i] = (_Float16)p3[i];
        }

        // B = V chunk 32x16 per DK half: half h -> V row m0+16g+h, col = lane%16
        // Rows >= N clamped: they multiply zeroed P columns -> contribute 0.
        float vf0[16], vf1[16];
        #pragma unroll
        for (int h = 0; h < 16; ++h) {
            const int m = imin(m0 + 16 * g + h, N_ - 1);
            vf0[h] = Vp[m * DK_ + ln];
            vf1[h] = Vp[m * DK_ + 16 + ln];
        }
        v16h bv0, bv1;
        #pragma unroll
        for (int h = 0; h < 16; ++h) {
            bv0[h] = (_Float16)vf0[h];
            bv1[h] = (_Float16)vf1[h];
        }

        acc0 = __builtin_amdgcn_wmma_f32_16x16x32_f16(false, ap, false, bv0,
                                                      (short)0, acc0, false, false);
        acc1 = __builtin_amdgcn_wmma_f32_16x16x32_f16(false, ap, false, bv1,
                                                      (short)0, acc1, false, false);
    }

    // ---- apply 1/sum per row, then store (f32 C/D layout) ----------------
    {
        const v4f i0 = *(const v4f*)(sinv + 8 * g);      // rows 8g..8g+3
        const v4f i1 = *(const v4f*)(sinv + 8 * g + 4);  // rows 8g+4..8g+7
        #pragma unroll
        for (int r = 0; r < 4; ++r) {
            acc0[r]     *= i0[r];
            acc1[r]     *= i0[r];
            acc0[4 + r] *= i1[r];
            acc1[4 + r] *= i1[r];
        }
    }

    #pragma unroll
    for (int r = 0; r < 8; ++r) {
        const int grow = row0 + r + 8 * g;
        if (grow < N_) {
            Op[grow * DK_ + ln]      = acc0[r];
            Op[grow * DK_ + 16 + ln] = acc1[r];
        }
    }
}

extern "C" void kernel_launch(void* const* d_in, const int* in_sizes, int n_in,
                              void* d_out, int out_size, void* d_ws, size_t ws_size,
                              hipStream_t stream)
{
    (void)in_sizes; (void)n_in; (void)d_ws; (void)ws_size; (void)out_size;

    const float* Q    = (const float*)d_in[0];
    const float* K    = (const float*)d_in[1];
    const float* V    = (const float*)d_in[2];
    const float* bias = (const float*)d_in[3];
    float*       out  = (float*)d_out;

    const int nblocks = B_ * H_ * T_ * ROW_TILES;   // 32256 single-wave WGs
    attn_wmma_kernel<<<dim3(nblocks), dim3(32), 0, stream>>>(Q, K, V, bias, out);
}
